// GCN_S_44229573214372
// MI455X (gfx1250) — matrix operand order repaired and checked by reading, not compiled, not run on hardware
//
#include <hip/hip_runtime.h>

#define NN   8192   // nodes
#define INF  512    // in features
#define OUTF 256    // out features

typedef __attribute__((ext_vector_type(16))) __bf16 v16bf;
typedef __attribute__((ext_vector_type(8)))  float  v8f;
typedef __attribute__((ext_vector_type(4)))  float  v4f;

static __device__ __forceinline__ v8f wmma_bf16(v16bf a, v16bf b, v8f c) {
    return __builtin_amdgcn_wmma_f32_16x16x32_bf16(
        /*neg_a=*/false, a, /*neg_b=*/false, b,
        /*c_mod=*/(short)0, c, /*reuse_a=*/false, /*reuse_b=*/false);
}

// Raw f32 data backing one 16x32 bf16 A-fragment (contiguous lane layout:
// lane holds K=[k+16h,+16), h = lane>=16) -> 4x b128 loads.
struct ARaw { v4f r0, r1, r2, r3; };

static __device__ __forceinline__ ARaw load_a_raw(const float* __restrict__ row,
                                                  int k, int h) {
    ARaw a;
    a.r0 = *(const v4f*)(row + k + 16 * h);
    a.r1 = *(const v4f*)(row + k + 16 * h + 4);
    a.r2 = *(const v4f*)(row + k + 16 * h + 8);
    a.r3 = *(const v4f*)(row + k + 16 * h + 12);
    return a;
}

static __device__ __forceinline__ v16bf cvt_a(const ARaw& a) {
    v16bf r;
#pragma unroll
    for (int i = 0; i < 4; ++i) {
        r[i]      = (__bf16)a.r0[i];
        r[4 + i]  = (__bf16)a.r1[i];
        r[8 + i]  = (__bf16)a.r2[i];
        r[12 + i] = (__bf16)a.r3[i];
    }
    return r;
}

// 32x16 bf16 B-fragment: 16 contiguous K values of column n from the
// transposed panel (row n of the transposed buffer) -> one 32-byte load.
static __device__ __forceinline__ v16bf load_b_bf16(const __bf16* __restrict__ p) {
    return *(const v16bf*)p;
}

// ---------------------------------------------------------------------------
// Kernel 1: d[i] = rsqrt(1 + sum_j A[i,j]);  one block per row, b128 streaming
// ---------------------------------------------------------------------------
__global__ __launch_bounds__(256) void gcn_rowsum_d(const float* __restrict__ A,
                                                    float* __restrict__ d) {
    const int row = blockIdx.x;
    const v4f* p = (const v4f*)(A + (size_t)row * NN);
    float s = 0.0f;
#pragma unroll
    for (int i = 0; i < NN / 4 / 256; ++i) {
        v4f v = p[threadIdx.x + i * 256];
        s += v.x + v.y + v.z + v.w;
    }
    __shared__ float red[256];
    red[threadIdx.x] = s;
    __syncthreads();
#pragma unroll
    for (int off = 128; off > 0; off >>= 1) {
        if (threadIdx.x < off) red[threadIdx.x] += red[threadIdx.x + off];
        __syncthreads();
    }
    if (threadIdx.x == 0) d[row] = rsqrtf(1.0f + red[0]);
}

// ---------------------------------------------------------------------------
// Kernel 2: Wt[n,f] = bf16(W[f,n])   (256x512 bf16, tiny)
// ---------------------------------------------------------------------------
__global__ __launch_bounds__(256) void gcn_wt(const float* __restrict__ W,
                                              __bf16* __restrict__ Wt) {
    const int n = blockIdx.x;   // 0..255
    for (int f = threadIdx.x; f < INF; f += 256)
        Wt[(size_t)n * INF + f] = (__bf16)W[(size_t)f * OUTF + n];
}

// ---------------------------------------------------------------------------
// Kernel 3: SbT[n,k] = bf16(d[k] * (X@W)[k,n])  via bf16 WMMA, f32 accumulate.
// Wave -> 16 rows (nodes) x 64 cols; transposed C-store is one b128 per tile.
// Software-pipelined: next-K fragments are in flight during current WMMAs.
// ---------------------------------------------------------------------------
__global__ __launch_bounds__(256) void gcn_support(const float* __restrict__ X,
                                                   const __bf16* __restrict__ Wt,
                                                   const float* __restrict__ d,
                                                   __bf16* __restrict__ SbT) {
    const int lane  = threadIdx.x & 31;
    const int wave  = threadIdx.x >> 5;
    const int gw    = blockIdx.x * 8 + wave;
    const int mtile = gw >> 2;          // 0..511 over node rows
    const int ngrp  = gw & 3;           // 0..3  -> 64-col group
    const int lm = lane & 15, h = lane >> 4;

    const float* xrow = X + (size_t)(mtile * 16 + lm) * INF;
    const int n0 = ngrp * 64;

    const __bf16* bp[4];
#pragma unroll
    for (int t = 0; t < 4; ++t)
        bp[t] = Wt + (size_t)(n0 + t * 16 + lm) * INF + 16 * h;

    v8f acc[4] = {v8f{}, v8f{}, v8f{}, v8f{}};

    ARaw  ar = load_a_raw(xrow, 0, h);
    v16bf br[4];
#pragma unroll
    for (int t = 0; t < 4; ++t) br[t] = load_b_bf16(bp[t]);

    for (int k = 0; k < INF - 32; k += 32) {
        v16bf a = cvt_a(ar);
        v16bf b0 = br[0], b1 = br[1], b2 = br[2], b3 = br[3];
        ar = load_a_raw(xrow, k + 32, h);            // next-iteration loads
#pragma unroll
        for (int t = 0; t < 4; ++t) br[t] = load_b_bf16(bp[t] + k + 32);
        acc[0] = wmma_bf16(a, b0, acc[0]);
        acc[1] = wmma_bf16(a, b1, acc[1]);
        acc[2] = wmma_bf16(a, b2, acc[2]);
        acc[3] = wmma_bf16(a, b3, acc[3]);
    }
    {   // peeled last iteration
        v16bf a = cvt_a(ar);
#pragma unroll
        for (int t = 0; t < 4; ++t) acc[t] = wmma_bf16(a, br[t], acc[t]);
    }

    const int rowbase = mtile * 16 + h * 8;   // 8 consecutive node indices
    float scale[8];
#pragma unroll
    for (int v = 0; v < 8; ++v) scale[v] = d[rowbase + v];
#pragma unroll
    for (int t = 0; t < 4; ++t) {
        union { __bf16 hh[8]; uint4 u; } pk;
#pragma unroll
        for (int v = 0; v < 8; ++v) pk.hh[v] = (__bf16)(acc[t][v] * scale[v]);
        const int col = n0 + t * 16 + lm;
        *(uint4*)(SbT + (size_t)col * NN + rowbase) = pk.u;  // transposed b128 store
    }
}

// ---------------------------------------------------------------------------
// Kernel 4: out[m,n] = d[m] * sum_k A[m,k] * SbT[n,k]   (the 34.4-GFLOP GEMM)
// Wave -> 16x64 strip; A converted f32->bf16 in registers; SbT L2-resident.
// Double-buffered K loop: loads for K+32 overlap the 4 WMMAs + converts of K.
// ---------------------------------------------------------------------------
__global__ __launch_bounds__(256) void gcn_aggregate(const float* __restrict__ A,
                                                     const __bf16* __restrict__ SbT,
                                                     const float* __restrict__ d,
                                                     float* __restrict__ out) {
    const int lane  = threadIdx.x & 31;
    const int wave  = threadIdx.x >> 5;
    const int gw    = blockIdx.x * 8 + wave;
    const int mtile = gw >> 2;          // 0..511  (waves 0..3 share A rows)
    const int ngrp  = gw & 3;
    const int lm = lane & 15, h = lane >> 4;

    const float* arow = A + (size_t)(mtile * 16 + lm) * NN;
    const int n0 = ngrp * 64;

    const __bf16* bp[4];
#pragma unroll
    for (int t = 0; t < 4; ++t)
        bp[t] = SbT + (size_t)(n0 + t * 16 + lm) * NN + 16 * h;

    v8f acc[4] = {v8f{}, v8f{}, v8f{}, v8f{}};

    ARaw  ar = load_a_raw(arow, 0, h);
    v16bf br[4];
#pragma unroll
    for (int t = 0; t < 4; ++t) br[t] = load_b_bf16(bp[t]);

    for (int k = 0; k < NN - 32; k += 32) {
        __builtin_prefetch(arow + k + 1024, 0, 0);   // global_prefetch, A stream
        v16bf a = cvt_a(ar);
        v16bf b0 = br[0], b1 = br[1], b2 = br[2], b3 = br[3];
        ar = load_a_raw(arow, k + 32, h);            // next-iteration loads
#pragma unroll
        for (int t = 0; t < 4; ++t) br[t] = load_b_bf16(bp[t] + k + 32);
        acc[0] = wmma_bf16(a, b0, acc[0]);
        acc[1] = wmma_bf16(a, b1, acc[1]);
        acc[2] = wmma_bf16(a, b2, acc[2]);
        acc[3] = wmma_bf16(a, b3, acc[3]);
    }
    {   // peeled last iteration
        v16bf a = cvt_a(ar);
#pragma unroll
        for (int t = 0; t < 4; ++t) acc[t] = wmma_bf16(a, br[t], acc[t]);
    }

    const int rowbase = mtile * 16 + h * 8;
#pragma unroll
    for (int v = 0; v < 8; ++v) {
        const float dm = d[rowbase + v];
        const size_t ro = (size_t)(rowbase + v) * OUTF;
#pragma unroll
        for (int t = 0; t < 4; ++t)
            out[ro + n0 + t * 16 + lm] = dm * acc[t][v];
    }
}

// ---------------------------------------------------------------------------
extern "C" void kernel_launch(void* const* d_in, const int* in_sizes, int n_in,
                              void* d_out, int out_size, void* d_ws, size_t ws_size,
                              hipStream_t stream) {
    const float* A = (const float*)d_in[0];   // 8192 x 8192
    const float* X = (const float*)d_in[1];   // 8192 x 512
    const float* W = (const float*)d_in[2];   // 512 x 256
    float* out = (float*)d_out;               // 8192 x 256

    char* ws = (char*)d_ws;
    float*  dvec = (float*)(ws);                                   // 32 KB
    __bf16* Wt   = (__bf16*)(ws + 32768);                          // 256 KB
    __bf16* SbT  = (__bf16*)(ws + 32768 + (size_t)OUTF * INF * 2); // 4 MB

    gcn_rowsum_d<<<NN, 256, 0, stream>>>(A, dvec);
    gcn_wt<<<OUTF, 256, 0, stream>>>(W, Wt);
    // 512 m-tiles * 4 n-groups = 2048 waves = 256 blocks of 8 waves
    gcn_support<<<(NN / 16) * 4 / 8, 256, 0, stream>>>(X, Wt, dvec, SbT);
    gcn_aggregate<<<(NN / 16) * 4 / 8, 256, 0, stream>>>(A, SbT, dvec, out);
}
